// PainnModel_9612136809069
// MI455X (gfx1250) — compile-verified, matrix operand order checked by reading, not run on hardware
//
#include <hip/hip_runtime.h>
#include <hip/hip_bf16.h>

// PaiNN forward for MI455X (gfx1250, wave32).
// GEMMs use v_wmma_f32_16x16x32_bf16 (bf16 in, fp32 accum). Edge message
// passing is a fused wave-per-edge kernel with f32 L2 atomics.

#define NATOMS  50000
#define NEDGES  800000
#define HDIM    128
#define KRBF    20
#define NLAYERS 3
#define PI_F    3.14159265358979323846f
#define CUT_F   5.0f

typedef __bf16 bf16_t;
typedef bf16_t        v16bf __attribute__((ext_vector_type(16)));
typedef float         v8f   __attribute__((ext_vector_type(8)));
typedef unsigned short v16us __attribute__((ext_vector_type(16)));
typedef unsigned short v8us  __attribute__((ext_vector_type(8)));

__device__ __forceinline__ unsigned short f2bf(float f) {
  unsigned int u = __float_as_uint(f);
  u += 0x7FFFu + ((u >> 16) & 1u);           // round-to-nearest-even
  return (unsigned short)(u >> 16);
}

__device__ __forceinline__ v16bf as_bf(v16us u) {
  union { v16us u; v16bf b; } c; c.u = u; return c.b;
}

__device__ __forceinline__ float silu_f(float x) {
  return x * (1.0f / (1.0f + __expf(-x)));
}

// ---------------------------------------------------------------------------
// Edge geometry: unit vectors + distances (rbf/fc recomputed later in LDS).
// ---------------------------------------------------------------------------
__global__ void edge_geom_kernel(const float* __restrict__ pos,
                                 const int* __restrict__ edges,
                                 float* __restrict__ unitv,
                                 float* __restrict__ dist) {
  int e = blockIdx.x * blockDim.x + threadIdx.x;
  if (e >= NEDGES) return;
  int s = edges[2 * e], r = edges[2 * e + 1];
  float dx = pos[3 * r + 0] - pos[3 * s + 0];
  float dy = pos[3 * r + 1] - pos[3 * s + 1];
  float dz = pos[3 * r + 2] - pos[3 * s + 2];
  float d  = sqrtf(dx * dx + dy * dy + dz * dz);
  float inv = 1.0f / fmaxf(d, 1e-12f);
  unitv[3 * e + 0] = dx * inv;
  unitv[3 * e + 1] = dy * inv;
  unitv[3 * e + 2] = dz * inv;
  dist[e] = d;
}

// ---------------------------------------------------------------------------
// State init: s = emb[atoms], v = 0.   (s,v live in d_out)
// ---------------------------------------------------------------------------
__global__ void init_state_kernel(const int* __restrict__ atoms,
                                  const float* __restrict__ emb,
                                  float* __restrict__ s,
                                  float* __restrict__ v) {
  int i = blockIdx.x * blockDim.x + threadIdx.x;
  if (i < NATOMS * HDIM) {
    int node = i >> 7, h = i & 127;
    s[i] = emb[atoms[node] * HDIM + h];
  }
  if (i < NATOMS * 3 * HDIM) v[i] = 0.0f;
}

__global__ void f2bf_kernel(const float* __restrict__ src,
                            unsigned short* __restrict__ dst, int n) {
  int i = blockIdx.x * blockDim.x + threadIdx.x;
  if (i < n) dst[i] = f2bf(src[i]);
}

__global__ void zero_kernel(float* __restrict__ p, int n) {
  int i = blockIdx.x * blockDim.x + threadIdx.x;
  if (i < n) p[i] = 0.0f;
}

__global__ void add_inplace_kernel(float* __restrict__ dst,
                                   const float* __restrict__ src, int n) {
  int i = blockIdx.x * blockDim.x + threadIdx.x;
  if (i < n) dst[i] += src[i];
}

// ---------------------------------------------------------------------------
// Pack a K x Nc f32 weight matrix into the WMMA bf16 B-operand lane layout:
// P[((ntile*ktiles + ktile)*32 + lane)*16 + j]
//   lane 0-15  : n = ntile*16+lane, j<8 -> K=kt*32+j,     j>=8 -> K=kt*32+8+j
//   lane 16-31 : n = ntile*16+(lane&15), kgrp = 8 offset (per ISA 16-bit layout)
// ---------------------------------------------------------------------------
__global__ void pack_b_kernel(const float* __restrict__ W,
                              unsigned short* __restrict__ P,
                              int K, int Nc) {
  int idx = blockIdx.x * blockDim.x + threadIdx.x;
  if (idx >= K * Nc) return;
  int j    = idx & 15;
  int lane = (idx >> 4) & 31;
  int t    = idx >> 9;                 // ntile*ktiles + ktile
  int ktiles = K >> 5;
  int ntile = t / ktiles;
  int ktile = t - ntile * ktiles;
  int n    = ntile * 16 + (lane & 15);
  int kgrp = (lane >> 4) * 8;
  int k    = ktile * 32 + kgrp + ((j < 8) ? j : (8 + j));  // j>=8 => +16+(j-8)
  P[idx] = f2bf(W[k * Nc + n]);
}

// ---------------------------------------------------------------------------
// WMMA bf16 GEMM:  C[M x Nc] = act(A[M x K] @ B + bias)
// A row-major bf16 (lda=K); B pre-packed (pack_b_kernel); one 16x16 C tile
// per wave32; 8 waves per block.  M%16==0, K%32==0, Nc%16==0.
// ---------------------------------------------------------------------------
template <int ACT, int OUTBF>
__global__ void __launch_bounds__(256)
gemm_bf16_kernel(const unsigned short* __restrict__ A,
                 const unsigned short* __restrict__ Bp,
                 const float* __restrict__ bias,
                 float* __restrict__ Cf,
                 unsigned short* __restrict__ Cb,
                 int M, int K, int Nc) {
  const int lane = threadIdx.x & 31;
  const int wave = threadIdx.x >> 5;
  const int ntiles = Nc >> 4;
  const int ktiles = K >> 5;
  int gw = blockIdx.x * 8 + wave;
  int totalTiles = (M >> 4) * ntiles;
  if (gw >= totalTiles) return;
  int tile_m = gw / ntiles;
  int tile_n = gw - tile_m * ntiles;

  const long mrow = (long)tile_m * 16 + (lane & 15);
  const int  kgrp = (lane >> 4) * 8;
  const unsigned short* Arow = A + mrow * K + kgrp;
  const v16us* Bt = ((const v16us*)Bp) + (long)tile_n * ktiles * 32 + lane;

  v8f acc = {};
  for (int kt = 0; kt < ktiles; ++kt) {
    v8us alo = *(const v8us*)(Arow + kt * 32);
    v8us ahi = *(const v8us*)(Arow + kt * 32 + 16);
    v16us au;
#pragma unroll
    for (int j = 0; j < 8; ++j) { au[j] = alo[j]; au[8 + j] = ahi[j]; }
    v16us bu = Bt[(long)kt * 32];
    acc = __builtin_amdgcn_wmma_f32_16x16x32_bf16(
        false, as_bf(au), false, as_bf(bu), (short)0, acc, false, false);
  }

  const int coln = tile_n * 16 + (lane & 15);
  const float bval = bias ? bias[coln] : 0.0f;
  const int rbase = (lane >> 4) * 8;
#pragma unroll
  for (int r = 0; r < 8; ++r) {
    long mm = (long)tile_m * 16 + rbase + r;
    float val = acc[r] + bval;
    if (ACT) val = silu_f(val);
    if (OUTBF) Cb[mm * Nc + coln] = f2bf(val);
    else       Cf[mm * Nc + coln] = val;
  }
}

// ---------------------------------------------------------------------------
// Fused edge message pass. One wave per edge (8 edges / 256-thread block).
// rbf/fc for the block's 8 edges computed once into LDS. filt = rbf@filt_W
// fused in-register (weights L2-resident). Scatter: s via atomics into s,
// v-messages via atomics into dv (so v[sender] reads see the OLD v).
// ---------------------------------------------------------------------------
__global__ void __launch_bounds__(256)
message_kernel(const int* __restrict__ edges,
               const float* __restrict__ dist,
               const float* __restrict__ unitv,
               const float* __restrict__ phi,
               const float* __restrict__ filt_W,   // 20 x 384 (layer slice)
               const float* __restrict__ filt_b,   // 384
               const float* __restrict__ v,        // N x 3 x 128 (old)
               float* __restrict__ s,              // N x 128 (accumulated)
               float* __restrict__ dv) {           // N x 3 x 128 (delta)
  __shared__ float rbf_s[8][KRBF];
  __shared__ float fc_s[8];

  int e0 = blockIdx.x * 8;
  int t  = threadIdx.x;
  if (t < 8 * KRBF) {
    int le = t / KRBF, k = t - le * KRBF;
    int e = e0 + le;
    if (e < NEDGES) {
      float d  = dist[e];
      float dn = fmaxf(d, 1e-12f);
      rbf_s[le][k] = __sinf(d * ((float)(k + 1) * (PI_F / CUT_F))) / dn;
      if (k == 0)
        fc_s[le] = (d < CUT_F) ? 0.5f * (__cosf(PI_F * d * (1.0f / CUT_F)) + 1.0f)
                               : 0.0f;
    }
  }
  __syncthreads();

  int wave = threadIdx.x >> 5;
  int lane = threadIdx.x & 31;
  int e = e0 + wave;
  if (e >= NEDGES) return;

  int snd = edges[2 * e], rcv = edges[2 * e + 1];
  float fc = fc_s[wave];
  float ux = unitv[3 * e + 0], uy = unitv[3 * e + 1], uz = unitv[3 * e + 2];
  const float* rb   = rbf_s[wave];
  const float* phis = phi + (long)snd * 384;
  const float* vs   = v   + (long)snd * 384;
  float* sr = s  + (long)rcv * 128;
  float* vr = dv + (long)rcv * 384;

#pragma unroll
  for (int j = 0; j < 4; ++j) {
    int h = j * 32 + lane;
    float fg = filt_b[h], fe = filt_b[128 + h], fm = filt_b[256 + h];
#pragma unroll
    for (int k = 0; k < KRBF; ++k) {
      float r = rb[k];
      const float* wrow = filt_W + k * 384;
      fg = fmaf(r, wrow[h],       fg);
      fe = fmaf(r, wrow[128 + h], fe);
      fm = fmaf(r, wrow[256 + h], fm);
    }
    fg *= fc; fe *= fc; fm *= fc;
    float gv = fg * phis[h];         // gate_v
    float ge = fe * phis[128 + h];   // gate_e
    float ms = fm * phis[256 + h];   // msg_s
    atomicAdd(&sr[h], ms);
    atomicAdd(&vr[h],       fmaf(vs[h],       gv, ge * ux));
    atomicAdd(&vr[128 + h], fmaf(vs[128 + h], gv, ge * uy));
    atomicAdd(&vr[256 + h], fmaf(vs[256 + h], gv, ge * uz));
  }
}

// ---------------------------------------------------------------------------
// Vn = sqrt(sum_d Vv^2 + eps);  x_bf = concat(s, Vn) in bf16 for the update MLP
// ---------------------------------------------------------------------------
__global__ void vn_concat_kernel(const float* __restrict__ s,
                                 const float* __restrict__ Vv,
                                 unsigned short* __restrict__ x_bf) {
  int i = blockIdx.x * blockDim.x + threadIdx.x;
  if (i >= NATOMS * HDIM) return;
  int node = i >> 7, h = i & 127;
  const float* vv = Vv + (long)node * 384;
  float vn = sqrtf(vv[h] * vv[h] + vv[128 + h] * vv[128 + h] +
                   vv[256 + h] * vv[256 + h] + 1e-12f);
  x_bf[(long)node * 256 + h]       = f2bf(s[i]);
  x_bf[(long)node * 256 + 128 + h] = f2bf(vn);
}

// ---------------------------------------------------------------------------
// Final per-node update: inner = <Uv,Vv>; s += a_ss + a_sv*inner; v += a_vv*Uv
// ---------------------------------------------------------------------------
__global__ void update_kernel(const float* __restrict__ a,
                              const float* __restrict__ Uv,
                              const float* __restrict__ Vv,
                              float* __restrict__ s,
                              float* __restrict__ v) {
  int i = blockIdx.x * blockDim.x + threadIdx.x;
  if (i >= NATOMS * HDIM) return;
  int node = i >> 7, h = i & 127;
  const float* u  = Uv + (long)node * 384;
  const float* w  = Vv + (long)node * 384;
  const float* an = a  + (long)node * 384;
  float inner = u[h] * w[h] + u[128 + h] * w[128 + h] + u[256 + h] * w[256 + h];
  s[i] += an[h] + an[128 + h] * inner;
  float avv = an[256 + h];
  float* vn = v + (long)node * 384;
  vn[h]       += avv * u[h];
  vn[128 + h] += avv * u[128 + h];
  vn[256 + h] += avv * u[256 + h];
}

// ---------------------------------------------------------------------------
// Host orchestration
// ---------------------------------------------------------------------------
extern "C" void kernel_launch(void* const* d_in, const int* in_sizes, int n_in,
                              void* d_out, int out_size, void* d_ws, size_t ws_size,
                              hipStream_t stream) {
  const int*   atoms   = (const int*)d_in[0];
  const float* pos     = (const float*)d_in[1];
  const int*   edges   = (const int*)d_in[2];
  const float* emb     = (const float*)d_in[3];
  const float* filt_W  = (const float*)d_in[4];
  const float* filt_b  = (const float*)d_in[5];
  const float* smlp_W1 = (const float*)d_in[6];
  const float* smlp_b1 = (const float*)d_in[7];
  const float* smlp_W2 = (const float*)d_in[8];
  const float* smlp_b2 = (const float*)d_in[9];
  const float* U_W     = (const float*)d_in[10];
  const float* V_W     = (const float*)d_in[11];
  const float* umlp_W1 = (const float*)d_in[12];
  const float* umlp_b1 = (const float*)d_in[13];
  const float* umlp_W2 = (const float*)d_in[14];
  const float* umlp_b2 = (const float*)d_in[15];

  float* s_state = (float*)d_out;                       // N x 128
  float* v_state = s_state + (size_t)NATOMS * HDIM;     // N x 3 x 128

  // workspace carve-out (256B aligned)
  char* base = (char*)d_ws;
  size_t off = 0;
  auto take = [&](size_t bytes) -> void* {
    void* p = base + off;
    off = (off + bytes + 255) & ~(size_t)255;
    return p;
  };
  float*          unitv  = (float*)take((size_t)NEDGES * 3 * 4);
  float*          distb  = (float*)take((size_t)NEDGES * 4);
  float*          phi_a  = (float*)take((size_t)NATOMS * 384 * 4);  // phi, later a
  float*          Uv     = (float*)take((size_t)NATOMS * 384 * 4);  // also dv scratch
  float*          Vv     = (float*)take((size_t)NATOMS * 384 * 4);
  unsigned short* s_bf   = (unsigned short*)take((size_t)NATOMS * 128 * 2);
  unsigned short* v_bf   = (unsigned short*)take((size_t)NATOMS * 384 * 2);
  unsigned short* hid_bf = (unsigned short*)take((size_t)NATOMS * 128 * 2);  // also a1
  unsigned short* x_bf   = (unsigned short*)take((size_t)NATOMS * 256 * 2);
  unsigned short* w1p    = (unsigned short*)take(128 * 128 * 2);
  unsigned short* w2p    = (unsigned short*)take(128 * 384 * 2);
  unsigned short* up     = (unsigned short*)take(128 * 128 * 2);
  unsigned short* vp     = (unsigned short*)take(128 * 128 * 2);
  unsigned short* uw1p   = (unsigned short*)take(256 * 128 * 2);
  unsigned short* uw2p   = (unsigned short*)take(128 * 384 * 2);
  float* dv = Uv;  // Uv buffer is dead during the message phase

  const int T = 256;
  auto cdiv = [](long a, long b) { return (int)((a + b - 1) / b); };

  edge_geom_kernel<<<cdiv(NEDGES, T), T, 0, stream>>>(pos, edges, unitv, distb);
  init_state_kernel<<<cdiv((long)NATOMS * 384, T), T, 0, stream>>>(atoms, emb,
                                                                   s_state, v_state);

  for (int i = 0; i < NLAYERS; ++i) {
    // pack layer weights into WMMA B layout (bf16)
    pack_b_kernel<<<cdiv(128 * 128, T), T, 0, stream>>>(smlp_W1 + (size_t)i * 128 * 128, w1p, 128, 128);
    pack_b_kernel<<<cdiv(128 * 384, T), T, 0, stream>>>(smlp_W2 + (size_t)i * 128 * 384, w2p, 128, 384);
    pack_b_kernel<<<cdiv(128 * 128, T), T, 0, stream>>>(U_W     + (size_t)i * 128 * 128, up,  128, 128);
    pack_b_kernel<<<cdiv(128 * 128, T), T, 0, stream>>>(V_W     + (size_t)i * 128 * 128, vp,  128, 128);
    pack_b_kernel<<<cdiv(256 * 128, T), T, 0, stream>>>(umlp_W1 + (size_t)i * 256 * 128, uw1p, 256, 128);
    pack_b_kernel<<<cdiv(128 * 384, T), T, 0, stream>>>(umlp_W2 + (size_t)i * 128 * 384, uw2p, 128, 384);

    // phi = silu(s@W1+b1)@W2+b2
    f2bf_kernel<<<cdiv((long)NATOMS * 128, T), T, 0, stream>>>(s_state, s_bf, NATOMS * 128);
    {
      int tiles1 = (NATOMS / 16) * (128 / 16);
      gemm_bf16_kernel<1, 1><<<cdiv(tiles1, 8), T, 0, stream>>>(
          s_bf, w1p, smlp_b1 + (size_t)i * 128, nullptr, hid_bf, NATOMS, 128, 128);
      int tiles2 = (NATOMS / 16) * (384 / 16);
      gemm_bf16_kernel<0, 0><<<cdiv(tiles2, 8), T, 0, stream>>>(
          hid_bf, w2p, smlp_b2 + (size_t)i * 384, phi_a, nullptr, NATOMS, 128, 384);
    }

    // edge messages -> s (atomics), dv (atomics); then v += dv
    zero_kernel<<<cdiv((long)NATOMS * 384, T), T, 0, stream>>>(dv, NATOMS * 384);
    message_kernel<<<NEDGES / 8, T, 0, stream>>>(
        edges, distb, unitv, phi_a, filt_W + (size_t)i * KRBF * 384,
        filt_b + (size_t)i * 384, v_state, s_state, dv);
    add_inplace_kernel<<<cdiv((long)NATOMS * 384, T), T, 0, stream>>>(
        v_state, dv, NATOMS * 384);

    // Uv = v@U, Vv = v@V   (M = 3N)
    f2bf_kernel<<<cdiv((long)NATOMS * 384, T), T, 0, stream>>>(v_state, v_bf, NATOMS * 384);
    {
      int tiles = (3 * NATOMS / 16) * (128 / 16);
      gemm_bf16_kernel<0, 0><<<cdiv(tiles, 8), T, 0, stream>>>(
          v_bf, up, nullptr, Uv, nullptr, 3 * NATOMS, 128, 128);
      gemm_bf16_kernel<0, 0><<<cdiv(tiles, 8), T, 0, stream>>>(
          v_bf, vp, nullptr, Vv, nullptr, 3 * NATOMS, 128, 128);
    }

    // a = silu([s,Vn]@uW1+b1)@uW2+b2
    vn_concat_kernel<<<cdiv((long)NATOMS * 128, T), T, 0, stream>>>(s_state, Vv, x_bf);
    {
      int tiles1 = (NATOMS / 16) * (128 / 16);
      gemm_bf16_kernel<1, 1><<<cdiv(tiles1, 8), T, 0, stream>>>(
          x_bf, uw1p, umlp_b1 + (size_t)i * 128, nullptr, hid_bf, NATOMS, 256, 128);
      int tiles2 = (NATOMS / 16) * (384 / 16);
      gemm_bf16_kernel<0, 0><<<cdiv(tiles2, 8), T, 0, stream>>>(
          hid_bf, uw2p, umlp_b2 + (size_t)i * 384, phi_a, nullptr, NATOMS, 128, 384);
    }

    update_kernel<<<cdiv((long)NATOMS * 128, T), T, 0, stream>>>(
        phi_a, Uv, Vv, s_state, v_state);
  }
  (void)in_sizes; (void)n_in; (void)out_size; (void)ws_size;
}
